// Attention_Critic_81698867904832
// MI455X (gfx1250) — matrix operand order repaired and checked by reading, not compiled, not run on hardware
//
#include <hip/hip_runtime.h>
#include <hip/hip_bf16.h>

// ---------------------------------------------------------------------------
// Attention-Critic fused pipeline for MI455X (gfx1250, wave32, WMMA 16x16x32)
//   N_AGENTS=8, BATCH=16384, SDIM=64, ADIM=16, HID=128, HEADS=4, ATT_D=32
// All GEMMs run on v_wmma_f32_16x16x32_f16 with fp32 accumulation.
// ---------------------------------------------------------------------------

#define NA    8
#define BATCHN 16384
#define SDIM  64
#define ADIM  16
#define FDIM  80          // SDIM + ADIM
#define HID   128
#define HEADS 4
#define ATTD  32

typedef __attribute__((ext_vector_type(16))) _Float16 v16h;
typedef __attribute__((ext_vector_type(8)))  _Float16 v8h;
typedef __attribute__((ext_vector_type(8)))  float    v8f;

static __device__ __forceinline__ v8f wmma16(v16h a, v16h b, v8f c) {
  // D = A(16x32 f16) x B(32x16 f16) + C(16x16 f32)
  return __builtin_amdgcn_wmma_f32_16x16x32_f16(false, a, false, b, (short)0, c,
                                                false, false);
}

// A-fragment (16x32 f16) from an LDS row-major tile.
// Lane l: m = l&15, g = l>>4 ; halves 0..7 = K[g*8 .. g*8+7],
//                              halves 8..15 = K[16+g*8 .. 16+g*8+7]
static __device__ __forceinline__ v16h ld_a(const _Float16* row0, int rowStride,
                                            int colBase, int lane) {
  int m = lane & 15, g = lane >> 4;
  const _Float16* p = row0 + m * rowStride + colBase + g * 8;
  union { v16h v; v8h h[2]; } u;
  u.h[0] = *(const v8h*)p;
  u.h[1] = *(const v8h*)(p + 16);
  return u.v;
}

// B-fragment (32x16 f16) pre-packed in global memory: frag[tile][lane][16]
static __device__ __forceinline__ v16h ld_b(const _Float16* frags, int tile,
                                            int lane) {
  const _Float16* p = frags + (size_t)tile * 512 + lane * 16;
  union { v16h v; uint4 q[2]; } u;
  u.q[0] = *(const uint4*)p;
  u.q[1] = *(const uint4*)(p + 8);
  return u.v;
}

static __device__ __forceinline__ float lrelu(float x) {
  return fmaxf(x, 0.01f * x);
}

// ---------------------------------------------------------------------------
// Kernel 1: BatchNorm stats. One block per (agent, feature); feature<64 -> s,
// else -> a. Biased variance, matching the reference.
// ---------------------------------------------------------------------------
__global__ void bn_stats(const float* __restrict__ s, const float* __restrict__ a,
                         float* __restrict__ meanArr, float* __restrict__ rstdArr) {
  int n = blockIdx.x / FDIM, f = blockIdx.x % FDIM;
  const float* src; int stride;
  if (f < SDIM) { src = s + (size_t)n * BATCHN * SDIM + f;          stride = SDIM; }
  else          { src = a + (size_t)n * BATCHN * ADIM + (f - SDIM); stride = ADIM; }
  float sum = 0.f, sq = 0.f;
  for (int b = threadIdx.x; b < BATCHN; b += 256) {
    float v = src[(size_t)b * stride];
    sum += v; sq += v * v;
  }
  __shared__ float s1[256], s2[256];
  s1[threadIdx.x] = sum; s2[threadIdx.x] = sq;
  __syncthreads();
  for (int off = 128; off > 0; off >>= 1) {
    if (threadIdx.x < off) {
      s1[threadIdx.x] += s1[threadIdx.x + off];
      s2[threadIdx.x] += s2[threadIdx.x + off];
    }
    __syncthreads();
  }
  if (threadIdx.x == 0) {
    float m = s1[0] * (1.0f / BATCHN);
    float var = s2[0] * (1.0f / BATCHN) - m * m;
    meanArr[blockIdx.x] = m;
    rstdArr[blockIdx.x] = rsqrtf(var + 1e-5f);
  }
}

// ---------------------------------------------------------------------------
// Kernel 2: pack an fp32 weight matrix set into f16 WMMA B-fragment layout.
// src element (mat, k, n) at: mat*matStride + k*kStride + (n%colMod) +
//                             (n/colMod)*blockStride   (covers the [P][H][D] case)
// One 32-lane block per 32x16 tile; lane writes its 16 contiguous f16.
// ---------------------------------------------------------------------------
__global__ void pack_w(const float* __restrict__ src, _Float16* __restrict__ dst,
                       int K, int KT, int NT, int matStride, int kStride,
                       int colMod, int blockStride) {
  int tile = blockIdx.x;
  int lane = threadIdx.x;
  int g = lane >> 4, nl = lane & 15;
  int nt = tile % NT, kt = (tile / NT) % KT, mt = tile / (NT * KT);
  _Float16* out = dst + (size_t)tile * 512 + lane * 16;
#pragma unroll
  for (int idx = 0; idx < 16; ++idx) {
    int k = kt * 32 + (idx < 8 ? g * 8 + idx : 16 + g * 8 + (idx - 8));
    int ncol = nt * 16 + nl;
    float v = 0.f;
    if (k < K)
      v = src[(size_t)mt * matStride + (size_t)k * kStride + (ncol % colMod) +
              (ncol / colMod) * blockStride];
    out[idx] = (_Float16)v;
  }
}

// ---------------------------------------------------------------------------
// Kernel 3: fully fused encoder -> projections -> attention -> critic.
// 1 block = 8 wave32s = all 8 agents for a 16-row batch tile. 160KB LDS.
// sA[w] cols 0..127: se -> h ; cols 128..255: sa_norm -> e -> other -> all_q
// ---------------------------------------------------------------------------
__global__ __launch_bounds__(256) void fused_ac(
    const float* __restrict__ s, const float* __restrict__ a,
    const float* __restrict__ b_sa, const float* __restrict__ b_se,
    const float* __restrict__ bv, const float* __restrict__ bc1,
    const float* __restrict__ bc2,
    const float* __restrict__ meanArr, const float* __restrict__ rstdArr,
    const _Float16* __restrict__ pWsa, const _Float16* __restrict__ pWse,
    const _Float16* __restrict__ pWk, const _Float16* __restrict__ pWsel,
    const _Float16* __restrict__ pWv, const _Float16* __restrict__ pWc1,
    const _Float16* __restrict__ pWc2, float* __restrict__ out) {
  __shared__ _Float16 sA[NA][16][256];    // 64KB
  __shared__ _Float16 sK[NA][16][HID];    // 32KB
  __shared__ _Float16 sV[NA][16][HID];    // 32KB
  __shared__ _Float16 sSel[NA][16][HID];  // 32KB

  const int w = threadIdx.x >> 5;  // wave id == agent id
  const int lane = threadIdx.x & 31;
  const int n = w;
  const int b0 = blockIdx.x * 16;
  _Float16* sAw = &sA[w][0][0];
  const int g = lane >> 4, nl = lane & 15;

  // ---- Stage 1: normalized concat(s,a) -> sA[w][:,128:224] (k-padded to 96)
  {
    const float* mA = meanArr + n * FDIM;
    const float* rA = rstdArr + n * FDIM;
    for (int t = lane; t < 16 * 96; t += 32) {
      int row = t / 96, col = t % 96;
      int b = b0 + row;
      float v = 0.f;
      if (col < SDIM)
        v = (s[((size_t)n * BATCHN + b) * SDIM + col] - mA[col]) * rA[col];
      else if (col < FDIM)
        v = (a[((size_t)n * BATCHN + b) * ADIM + (col - SDIM)] - mA[col]) * rA[col];
      sAw[row * 256 + 128 + col] = (_Float16)v;
    }
  }

  // ---- Stage 2: e = lrelu(sa@Wsa + b_sa) ; se = lrelu(s_n@Wse + b_se)
  {
    v8f accE[8] = {};
    v8f accS[8] = {};
#pragma unroll
    for (int kt = 0; kt < 3; ++kt) {
      v16h A = ld_a(sAw, 256, 128 + kt * 32, lane);
#pragma unroll
      for (int nt = 0; nt < 8; ++nt)
        accE[nt] = wmma16(A, ld_b(pWsa, (n * 3 + kt) * 8 + nt, lane), accE[nt]);
      if (kt < 2) {
#pragma unroll
        for (int nt = 0; nt < 8; ++nt)
          accS[nt] = wmma16(A, ld_b(pWse, (n * 2 + kt) * 8 + nt, lane), accS[nt]);
      }
    }
#pragma unroll
    for (int nt = 0; nt < 8; ++nt) {
      float be = b_sa[n * HID + nt * 16 + nl];
      float bs = b_se[n * HID + nt * 16 + nl];
#pragma unroll
      for (int r = 0; r < 8; ++r) {
        int row = 8 * g + r;
        sAw[row * 256 + nt * 16 + nl] = (_Float16)lrelu(accS[nt][r] + bs);        // se
        sAw[row * 256 + 128 + nt * 16 + nl] = (_Float16)lrelu(accE[nt][r] + be);  // e
      }
    }
  }

  // ---- Stage 3: keys = e@Wk ; vals = lrelu(e@Wv+bv) ; sel = se@Wsel
  {
    v16h Ae[4], As[4];
#pragma unroll
    for (int kt = 0; kt < 4; ++kt) {
      Ae[kt] = ld_a(sAw, 256, 128 + kt * 32, lane);
      As[kt] = ld_a(sAw, 256, kt * 32, lane);
    }
#pragma unroll
    for (int nt = 0; nt < 8; ++nt) {
      v8f ak = {}, av = {}, asel = {};
#pragma unroll
      for (int kt = 0; kt < 4; ++kt) {
        ak   = wmma16(Ae[kt], ld_b(pWk,   kt * 8 + nt, lane), ak);
        av   = wmma16(Ae[kt], ld_b(pWv,   kt * 8 + nt, lane), av);
        asel = wmma16(As[kt], ld_b(pWsel, kt * 8 + nt, lane), asel);
      }
      float bvv = bv[nt * 16 + nl];
#pragma unroll
      for (int r = 0; r < 8; ++r) {
        int row = 8 * g + r;
        sK[w][row][nt * 16 + nl]   = (_Float16)ak[r];
        sV[w][row][nt * 16 + nl]   = (_Float16)lrelu(av[r] + bvv);
        sSel[w][row][nt * 16 + nl] = (_Float16)asel[r];
      }
    }
  }
  __syncthreads();

  // ---- Stage 4: per-sample attention (fp32 VALU; 8x8xD is too small for WMMA)
  // lane -> (head p, agent i); wave handles rows w and w+8.
  {
    const int p = lane >> 3, i = lane & 7;
#pragma unroll
    for (int mi = 0; mi < 2; ++mi) {
      int m = w + 8 * mi;
      float selv[ATTD];
#pragma unroll
      for (int c = 0; c < 4; ++c) {
        v8h hv = *(const v8h*)&sSel[i][m][p * ATTD + c * 8];
#pragma unroll
        for (int t = 0; t < 8; ++t) selv[c * 8 + t] = (float)hv[t];
      }
      float logit[NA];
#pragma unroll
      for (int j = 0; j < NA; ++j) {
        float acc = 0.f;
#pragma unroll
        for (int c = 0; c < 4; ++c) {
          v8h hv = *(const v8h*)&sK[j][m][p * ATTD + c * 8];
#pragma unroll
          for (int t = 0; t < 8; ++t) acc += selv[c * 8 + t] * (float)hv[t];
        }
        logit[j] = (j == i) ? -1e9f : acc * 0.17677669529663687f;  // 1/sqrt(32)
      }
      float mx = logit[0];
#pragma unroll
      for (int j = 1; j < NA; ++j) mx = fmaxf(mx, logit[j]);
      float sum = 0.f;
#pragma unroll
      for (int j = 0; j < NA; ++j) { logit[j] = __expf(logit[j] - mx); sum += logit[j]; }
      float inv = 1.f / sum;
      float oth[ATTD];
#pragma unroll
      for (int d = 0; d < ATTD; ++d) oth[d] = 0.f;
#pragma unroll
      for (int j = 0; j < NA; ++j) {
        float wj = logit[j] * inv;
#pragma unroll
        for (int c = 0; c < 4; ++c) {
          v8h hv = *(const v8h*)&sV[j][m][p * ATTD + c * 8];
#pragma unroll
          for (int t = 0; t < 8; ++t) oth[c * 8 + t] += wj * (float)hv[t];
        }
      }
#pragma unroll
      for (int c = 0; c < 4; ++c) {
        v8h hv;
#pragma unroll
        for (int t = 0; t < 8; ++t) hv[t] = (_Float16)oth[c * 8 + t];
        *(v8h*)&sA[i][m][128 + p * ATTD + c * 8] = hv;  // other_flat col p*32+d
      }
    }
  }
  __syncthreads();

  // ---- Stage 5: critic MLP + argmax gather
  {
    v8f accH[8] = {};
#pragma unroll
    for (int kt = 0; kt < 8; ++kt) {  // K = 256 over concat(se, other)
      v16h A = ld_a(sAw, 256, kt * 32, lane);
#pragma unroll
      for (int nt = 0; nt < 8; ++nt)
        accH[nt] = wmma16(A, ld_b(pWc1, (n * 8 + kt) * 8 + nt, lane), accH[nt]);
    }
#pragma unroll
    for (int nt = 0; nt < 8; ++nt) {
      float bb = bc1[n * HID + nt * 16 + nl];
#pragma unroll
      for (int r = 0; r < 8; ++r)
        sAw[(8 * g + r) * 256 + nt * 16 + nl] = (_Float16)lrelu(accH[nt][r] + bb);  // h
    }
    v8f accQ = {};
#pragma unroll
    for (int kt = 0; kt < 4; ++kt) {
      v16h A = ld_a(sAw, 256, kt * 32, lane);
      accQ = wmma16(A, ld_b(pWc2, n * 4 + kt, lane), accQ);
    }
    float bq = bc2[n * ADIM + nl];
    float* qb = (float*)sAw;  // reuse sA[w] cols 128.. as fp32 all_q (row stride 128 floats)
#pragma unroll
    for (int r = 0; r < 8; ++r)
      qb[(8 * g + r) * 128 + 64 + nl] = accQ[r] + bq;

    if (lane < 16) {
      int b = b0 + lane;
      const float* ap = a + ((size_t)n * BATCHN + b) * ADIM;
      float best = ap[0]; int bi = 0;
#pragma unroll
      for (int k = 1; k < ADIM; ++k) {
        float v = ap[k];
        if (v > best) { best = v; bi = k; }
      }
      out[(size_t)n * BATCHN + b] = qb[lane * 128 + 64 + bi];
    }
  }
}

// ---------------------------------------------------------------------------
extern "C" void kernel_launch(void* const* d_in, const int* in_sizes, int n_in,
                              void* d_out, int out_size, void* d_ws, size_t ws_size,
                              hipStream_t stream) {
  (void)in_sizes; (void)n_in; (void)out_size; (void)ws_size;
  const float* s   = (const float*)d_in[0];
  const float* a   = (const float*)d_in[1];
  const float* Wsa = (const float*)d_in[2];
  const float* bsa = (const float*)d_in[3];
  const float* Wse = (const float*)d_in[4];
  const float* bse = (const float*)d_in[5];
  const float* Wk  = (const float*)d_in[6];
  const float* Wsl = (const float*)d_in[7];
  const float* Wv  = (const float*)d_in[8];
  const float* bv  = (const float*)d_in[9];
  const float* Wc1 = (const float*)d_in[10];
  const float* bc1 = (const float*)d_in[11];
  const float* Wc2 = (const float*)d_in[12];
  const float* bc2 = (const float*)d_in[13];

  char* ws = (char*)d_ws;
  float*    meanArr = (float*)(ws + 0);        // 8*80 f32
  float*    rstdArr = (float*)(ws + 2560);     // 8*80 f32
  _Float16* pWsa    = (_Float16*)(ws + 8192);    // 8*3*8*512 f16
  _Float16* pWse    = (_Float16*)(ws + 204800);  // 8*2*8*512
  _Float16* pWk     = (_Float16*)(ws + 335872);  // 4*8*512
  _Float16* pWsl    = (_Float16*)(ws + 368640);
  _Float16* pWv     = (_Float16*)(ws + 401408);
  _Float16* pWc1    = (_Float16*)(ws + 434176);  // 8*8*8*512
  _Float16* pWc2    = (_Float16*)(ws + 958464);  // 8*4*1*512  -> ~0.97MB total

  bn_stats<<<NA * FDIM, 256, 0, stream>>>(s, a, meanArr, rstdArr);
  // (K, KT, NT, matStride, kStride, colMod, blockStride)
  pack_w<<<8 * 3 * 8, 32, 0, stream>>>(Wsa, pWsa,  80, 3, 8, 80 * 128, 128, 128, 0);
  pack_w<<<8 * 2 * 8, 32, 0, stream>>>(Wse, pWse,  64, 2, 8, 64 * 128, 128, 128, 0);
  pack_w<<<4 * 8,     32, 0, stream>>>(Wk,  pWk,  128, 4, 8, 0, 32, 32, 128 * 32);
  pack_w<<<4 * 8,     32, 0, stream>>>(Wsl, pWsl, 128, 4, 8, 0, 32, 32, 128 * 32);
  pack_w<<<4 * 8,     32, 0, stream>>>(Wv,  pWv,  128, 4, 8, 0, 32, 32, 128 * 32);
  pack_w<<<8 * 8 * 8, 32, 0, stream>>>(Wc1, pWc1, 256, 8, 8, 256 * 128, 128, 128, 0);
  pack_w<<<8 * 4 * 1, 32, 0, stream>>>(Wc2, pWc2, 128, 4, 1, 128 * 16, 16, 16, 0);

  fused_ac<<<BATCHN / 16, 256, 0, stream>>>(s, a, bsa, bse, bv, bc1, bc2,
                                            meanArr, rstdArr, pWsa, pWse, pWk,
                                            pWsl, pWv, pWc1, pWc2, (float*)d_out);
}
